// CompressiveMemory_27230092657392
// MI455X (gfx1250) — matrix operand order repaired and checked by reading, not compiled, not run on hardware
//
#include <hip/hip_runtime.h>
#include <hip/hip_bf16.h>

// CompressiveMemory for MI455X (gfx1250, wave32, WMMA).
// B=128, H=64, D=64 fixed by the reference.
//
// retrieve: per (b,h) block: scores = MK^T x Q (f16 WMMA), /= (mnorm[d]+1e-6),
//           out = MV x scores (f16 WMMA). LDS-staged; B operands stored
//           column-major (stride 72 halves, 16B-aligned rows) so every WMMA
//           fragment is contiguous per lane -> ds_load_b128, not ds_load_u16.
// update:   ws-zero -> per-(b,h) partial sums with f32 global atomics -> finalize.

#define NB 128
#define NH 64
#define ND 64
#define LS 72   // LDS row stride in halves for transposed tiles (144B, 16B-aligned)

typedef __attribute__((ext_vector_type(16))) _Float16 v16h;
typedef __attribute__((ext_vector_type(8)))  float    v8f;

// One 16x16 tile of D[rm.., cn..] = A[rm.., 0..63] * B[0..63, cn..].
// Am: row-major A tile (stride lda halves). Bt: column-major B tile, i.e.
// Bt[col*ldb + k] == B[k][col]. Two v_wmma_f32_16x16x32_f16 cover K=64.
// Per-lane fragments use contiguous K = lg*16 + s for both A and B (wave32); all per-lane
// runs are contiguous so the compiler merges them into ds_load_b128.
__device__ __forceinline__ v8f tile_mma64(const _Float16* __restrict__ Am, int lda,
                                          const _Float16* __restrict__ Bt, int ldb,
                                          int rm, int cn, int lg, int l15)
{
    v8f acc = {};
    const _Float16* __restrict__ arow = Am + (rm + l15) * lda;
    const _Float16* __restrict__ bcol = Bt + (cn + l15) * ldb;
#pragma unroll
    for (int kk = 0; kk < 64; kk += 32) {
        v16h a, b;
        // A (16x32): VGPR v, lane-group lg -> K = lg*16 + v*2
#pragma unroll
        for (int v = 0; v < 8; ++v) {
            const int kb = lg * 16 + v * 2;
            a[2 * v]     = arow[kk + kb];
            a[2 * v + 1] = arow[kk + kb + 1];
        }
        // B (32x16): lanes 0-15 hold K=kk+0..15 of col l15, lanes 16-31 K=kk+16..31
        const int koff = kk + lg * 16;
#pragma unroll
        for (int e = 0; e < 16; ++e) b[e] = bcol[koff + e];
        acc = __builtin_amdgcn_wmma_f32_16x16x32_f16(
            /*neg_a=*/false, a, /*neg_b=*/false, b,
            /*c_mod=*/(short)0, acc, /*reuse_a=*/false, /*reuse_b=*/false);
    }
    return acc;
}

__global__ __launch_bounds__(256)
void retrieve_kernel(const float* __restrict__ q,      // [B,H,64,64]
                     const float* __restrict__ mk,     // [H,64,64]  (n,m)
                     const float* __restrict__ mnorm,  // [H]
                     const float* __restrict__ mv,     // [H,64,64]  (n,m)
                     float* __restrict__ out_mem,      // [B,H,64,64]
                     float* __restrict__ out_scores)   // [B,H,64,64]
{
    __shared__ _Float16 Qt [ND * LS];  // Q^T:   Qt[d*LS + n]  (B operand, GEMM 1)
    __shared__ _Float16 MKt[ND * LS];  // MK^T:  MKt[m*LS + n] (A operand, GEMM 1)
    __shared__ _Float16 MVs[ND * ND];  // MV:    MVs[n*64 + m] (A operand, GEMM 2)
    __shared__ _Float16 St [ND * LS];  // S^T:   St[d*LS + m]  (B operand, GEMM 2)

    const int bh  = blockIdx.x;            // b*H + h
    const int h   = bh & (NH - 1);
    const int tid = threadIdx.x;

    const float* __restrict__ qg  = q  + (size_t)bh * (ND * ND);
    const float* __restrict__ mkg = mk + (size_t)h  * (ND * ND);
    const float* __restrict__ mvg = mv + (size_t)h  * (ND * ND);

    // Stage tiles with b128 global loads; Q and MK transposed on the way in.
    for (int i = tid * 4; i < ND * ND; i += 256 * 4) {
        const int n = i >> 6, m0 = i & 63;          // i = n*64 + m0, m0 % 4 == 0
        const float4 qv = *(const float4*)(qg  + i);
        const float4 kv = *(const float4*)(mkg + i);
        const float4 vv = *(const float4*)(mvg + i);
        Qt [(m0 + 0) * LS + n] = (_Float16)qv.x;
        Qt [(m0 + 1) * LS + n] = (_Float16)qv.y;
        Qt [(m0 + 2) * LS + n] = (_Float16)qv.z;
        Qt [(m0 + 3) * LS + n] = (_Float16)qv.w;
        MKt[(m0 + 0) * LS + n] = (_Float16)kv.x;
        MKt[(m0 + 1) * LS + n] = (_Float16)kv.y;
        MKt[(m0 + 2) * LS + n] = (_Float16)kv.z;
        MKt[(m0 + 3) * LS + n] = (_Float16)kv.w;
        MVs[i + 0] = (_Float16)vv.x;
        MVs[i + 1] = (_Float16)vv.y;
        MVs[i + 2] = (_Float16)vv.z;
        MVs[i + 3] = (_Float16)vv.w;
    }
    __syncthreads();

    const int wave = tid >> 5;
    const int lane = tid & 31;
    const int lg   = lane >> 4;   // lane-group (0: lanes 0-15, 1: lanes 16-31)
    const int l15  = lane & 15;

    // GEMM 1: scores[m,d] = sum_n MK^T[m,n] * Q[n,d]; 16 tiles, 8 waves.
#pragma unroll
    for (int t = 0; t < 2; ++t) {
        const int tt = wave + t * 8;
        const int mi = tt >> 2, di = tt & 3;
        v8f acc = tile_mma64(MKt, LS, Qt, LS, mi * 16, di * 16, lg, l15);
        const int dcol = di * 16 + l15;               // trailing dim, fixed per lane
        const float inv = __builtin_amdgcn_rcpf(mnorm[dcol] + 1e-6f);
        float* __restrict__ so = out_scores + (size_t)bh * (ND * ND);
        const int m0 = mi * 16 + lg * 8;              // C/D layout: row = lg*8 + r
#pragma unroll
        for (int r = 0; r < 8; ++r) {
            const float s = acc[r] * inv;
            so[(m0 + r) * 64 + dcol] = s;
            St[dcol * LS + m0 + r] = (_Float16)s;     // 8 contiguous, 16B-aligned
        }
    }
    __syncthreads();

    // GEMM 2: out[n,d] = sum_m MV[n,m] * scores[m,d]
#pragma unroll
    for (int t = 0; t < 2; ++t) {
        const int tt = wave + t * 8;
        const int ni = tt >> 2, di = tt & 3;
        v8f acc = tile_mma64(MVs, ND, St, LS, ni * 16, di * 16, lg, l15);
        float* __restrict__ mo = out_mem + (size_t)bh * (ND * ND);
        const int n0 = ni * 16 + lg * 8;
#pragma unroll
        for (int r = 0; r < 8; ++r)
            mo[(n0 + r) * 64 + di * 16 + l15] = acc[r];
    }
}

// ---------------- update path ----------------

__global__ __launch_bounds__(256)
void u0_zero(float* __restrict__ ws, int n)
{
    const int i = blockIdx.x * 256 + threadIdx.x;
    if (i < n) ws[i] = 0.0f;
}

// Per (b,h): column sums of key/value over n, and partial binding
// pb[m] = sum_d ksum[d] * V[m][d]; accumulate into ws via f32 global atomics.
__global__ __launch_bounds__(256)
void u1_partial(const float* __restrict__ key, const float* __restrict__ val,
                float* __restrict__ ws_k,   // [H,64]
                float* __restrict__ ws_v,   // [H,64]
                float* __restrict__ ws_b)   // [64]
{
    __shared__ float Vs[ND * ND];
    __shared__ float kpart[4][ND];
    __shared__ float vpart[4][ND];
    __shared__ float ksum[ND];

    const int bh = blockIdx.x;
    const int h  = bh & (NH - 1);
    const float* __restrict__ kg = key + (size_t)bh * (ND * ND);
    const float* __restrict__ vg = val + (size_t)bh * (ND * ND);

    const int tid = threadIdx.x;
    const int c = tid & 63, r0 = tid >> 6;
    float ka = 0.0f, va = 0.0f;
    for (int n = r0; n < ND; n += 4) {
        const float v = vg[n * 64 + c];
        Vs[n * 64 + c] = v;
        va += v;
        ka += kg[n * 64 + c];
    }
    kpart[r0][c] = ka;
    vpart[r0][c] = va;
    __syncthreads();

    if (tid < 64) {
        const float ks = kpart[0][tid] + kpart[1][tid] + kpart[2][tid] + kpart[3][tid];
        const float vs = vpart[0][tid] + vpart[1][tid] + vpart[2][tid] + vpart[3][tid];
        ksum[tid] = ks;
        atomicAdd(&ws_k[h * 64 + tid], ks);
        atomicAdd(&ws_v[h * 64 + tid], vs);
    }
    __syncthreads();

    if (tid < 64) {
        const int m = tid;
        float pb = 0.0f;
        for (int d = 0; d < 64; ++d) pb += ksum[d] * Vs[m * 64 + d];
        atomicAdd(&ws_b[m], pb);
    }
}

__global__ __launch_bounds__(64)
void u2_final(const float* __restrict__ ws_k, const float* __restrict__ ws_v,
              const float* __restrict__ ws_b,
              const float* __restrict__ mnorm, const float* __restrict__ comp,
              float* __restrict__ out_key,   // [H,64]
              float* __restrict__ out_norm)  // [H]
{
    __shared__ float nmn_s[NH];
    const int h = threadIdx.x;                 // one thread per head
    const float invBN = 1.0f / (float)(NB * ND);

    float ss = 0.0f;
    for (int d = 0; d < 64; ++d) {
        const float km = ws_k[h * 64 + d] * invBN;  // mean key
        ss += km * km;
    }
    const float nmn = mnorm[h] + sqrtf(ss);
    nmn_s[h] = nmn;
    __syncthreads();

    float sum = 0.0f;
    for (int i = 0; i < NH; ++i) sum += nmn_s[i];
    const float factor = (sum * (1.0f / (float)NH) > 0.9f) ? comp[h] : 1.0f;

    out_norm[h] = nmn * factor;
    const float mb = ws_b[h] * (1.0f / (float)(NB * NH * ND));  // mean binding
    for (int d = 0; d < 64; ++d)
        out_key[h * 64 + d] = mb * (ws_v[h * 64 + d] * invBN) * factor;
}

extern "C" void kernel_launch(void* const* d_in, const int* in_sizes, int n_in,
                              void* d_out, int out_size, void* d_ws, size_t ws_size,
                              hipStream_t stream)
{
    const float* query = (const float*)d_in[0];
    const float* key   = (const float*)d_in[1];
    const float* value = (const float*)d_in[2];
    const float* mkey  = (const float*)d_in[3];
    const float* mnorm = (const float*)d_in[4];
    const float* mval  = (const float*)d_in[5];
    const float* comp  = (const float*)d_in[6];

    const size_t BHDD = (size_t)NB * NH * ND * ND;
    float* out_mem    = (float*)d_out;
    float* out_scores = out_mem + BHDD;
    float* out_mkey   = out_scores + BHDD;
    float* out_mnorm  = out_mkey + (size_t)NH * ND;

    float* ws_k = (float*)d_ws;              // [H,64]
    float* ws_v = ws_k + NH * ND;            // [H,64]
    float* ws_b = ws_v + NH * ND;            // [64]
    const int nz = NH * ND * 2 + ND;

    u0_zero   <<<(nz + 255) / 256, 256, 0, stream>>>((float*)d_ws, nz);
    u1_partial<<<NB * NH, 256, 0, stream>>>(key, value, ws_k, ws_v, ws_b);
    u2_final  <<<1, NH, 0, stream>>>(ws_k, ws_v, ws_b, mnorm, comp, out_mkey, out_mnorm);
    retrieve_kernel<<<NB * NH, 256, 0, stream>>>(query, mkey, mnorm, mval,
                                                 out_mem, out_scores);
}
